// VoxelHuman_1666447311481
// MI455X (gfx1250) — matrix operand order repaired and checked by reading, not compiled, not run on hardware
//
#include <hip/hip_runtime.h>
#include <hip/hip_bf16.h>

typedef __attribute__((ext_vector_type(16))) _Float16 v16h;
typedef __attribute__((ext_vector_type(8)))  _Float16 v8h;
typedef __attribute__((ext_vector_type(8)))  float    v8f;
typedef __attribute__((ext_vector_type(4)))  float    v4f;

#define BATCH   2
#define RAYS    8192
#define NSAMP   64
#define FEATC   148
#define UVS     65536
#define K0PAD   224          // 211-ch input padded to 7 * 32
#define HID     256
#define MTILE   64           // points per block
#define NPTS    (BATCH*RAYS*NSAMP)

// ---------------------------------------------------------------------------
// Fragment loaders (CDNA5 wave32 WMMA 16x16x32 f16 layouts, CDNA5 ISA)
// A 16x32: lane L holds M=L&15; halves e    -> K = (L>>4)*16 + e (contiguous order chosen by this code, not the ISA table order)
// B 32x16: lane L holds N=L&15; halves e    -> K = (L>>4)*16 + e (contiguous, same order as A)
// C 16x16 f32: vgpr v -> M = v + 8*(L>=16); N = L&15
// ---------------------------------------------------------------------------
__device__ __forceinline__ v16h load_afrag(const _Float16* base, int k0) {
  union { v16h v; v8h h[2]; } A;
  A.h[0] = *(const v8h*)(base + k0);
  A.h[1] = *(const v8h*)(base + k0 + 8);
  return A.v;
}
__device__ __forceinline__ v16h load_bfrag(const _Float16* p) {
  union { v16h v; v8h h[2]; } Bf;
  Bf.h[0] = *(const v8h*)(p);
  Bf.h[1] = *(const v8h*)(p + 8);
  return Bf.v;
}

__device__ __forceinline__ v8f wmma_f16(v16h a, v16h b, v8f c) {
  return __builtin_amdgcn_wmma_f32_16x16x32_f16(false, a, false, b, (short)0, c,
                                                false, false);
}

// 32x128 output tile per wave: 2 A-frags x 8 N-tiles, B double-buffered so the
// next weight fragment load overlaps the two WMMAs of the current one.
template<int KSTEPS, int KPAD>
__device__ __forceinline__ void mlp_layer(const _Float16* ar0, const _Float16* ar1,
                                          const _Float16* __restrict__ wbase,
                                          v8f acc0[8], v8f acc1[8])
{
  #pragma unroll
  for (int kk = 0; kk < KSTEPS; ++kk) {
    const v16h a0 = load_afrag(ar0, kk * 32);
    const v16h a1 = load_afrag(ar1, kk * 32);
    v16h bc = load_bfrag(wbase + kk * 32);
    #pragma unroll
    for (int nt = 0; nt < 8; ++nt) {
      v16h bn;
      if (nt < 7) bn = load_bfrag(wbase + (size_t)(nt + 1) * 16 * KPAD + kk * 32);
      else        bn = bc;
      acc0[nt] = wmma_f16(a0, bc, acc0[nt]);
      acc1[nt] = wmma_f16(a1, bc, acc1[nt]);
      bc = bn;
    }
  }
}

// ---------------------------------------------------------------------------
// Kernel 0: weights f32 -> f16, transposed [N][Kpad] so each B-fragment is one
// contiguous 32-byte run per lane.
// ---------------------------------------------------------------------------
__global__ __launch_bounds__(256)
void prep_weights(const float* __restrict__ w0, const float* __restrict__ w1,
                  const float* __restrict__ w_out,
                  _Float16* __restrict__ W0T, _Float16* __restrict__ W1T,
                  _Float16* __restrict__ WoutT)
{
  int g = blockIdx.x * 256 + threadIdx.x;
  const int T0 = 256 * K0PAD;          // 57344
  const int T1 = T0 + 256 * 256;       // +65536
  const int T2 = T1 + 16 * 256;        // +4096
  if (g < T0) {
    int n = g / K0PAD, kp = g % K0PAD;
    W0T[g] = (kp < 211) ? (_Float16)w0[kp * 256 + n] : (_Float16)0.0f;
  } else if (g < T1) {
    int i = g - T0; int n = i >> 8, k = i & 255;
    W1T[i] = (_Float16)w1[k * 256 + n];
  } else if (g < T2) {
    int i = g - T1; int n = i >> 8, k = i & 255;
    WoutT[i] = (n < 4) ? (_Float16)w_out[k * 4 + n] : (_Float16)0.0f;
  }
}

// ---------------------------------------------------------------------------
// Kernel 1: fused gather + embed + 3-layer MLP via WMMA. 64 points / block,
// 4 waves; waves tile the 64x256 hidden output as 2x2 blocks of 32x128.
// ---------------------------------------------------------------------------
__global__ __launch_bounds__(128)
void mlp_kernel(const float* __restrict__ rays_o, const float* __restrict__ rays_d,
                const float* __restrict__ uv_feat, const float* __restrict__ transforms,
                const float* __restrict__ b0, const float* __restrict__ b1,
                const float* __restrict__ b_out,
                const int* __restrict__ uv_idx, const int* __restrict__ vox_idx,
                const _Float16* __restrict__ W0T, const _Float16* __restrict__ W1T,
                const _Float16* __restrict__ WoutT, float* __restrict__ out4)
{
  __shared__ _Float16 smem[MTILE * HID];   // 32 KB: X[64][224] then H[64][256]
  __shared__ float bias0[HID];
  __shared__ float bias1[HID];

  const int tid   = threadIdx.x;
  const int lane  = tid & 31;
  const int wave  = tid >> 5;
  const int base  = blockIdx.x * MTILE;
  const int lm    = lane & 15;
  const int lg    = lane >> 4;
  const int mbase = (wave >> 1) * 32;      // rows  0 or 32
  const int nbase = (wave & 1) * 128;      // cols  0 or 128

  bias0[tid] = b0[tid]; bias0[tid + 128] = b0[tid + 128];
  bias1[tid] = b1[tid]; bias1[tid + 128] = b1[tid + 128];

  // ---- stage 0: build 211-ch f16 input rows in LDS (2 threads / point) ----
  {
    const int j = tid >> 1;            // point within tile
    const int half = tid & 1;
    const int p = base + j;
    const int b = p / (RAYS * NSAMP);
    const int rem = p - b * (RAYS * NSAMP);
    const int r = rem >> 6;
    const int s = rem & 63;
    _Float16* row = &smem[j * K0PAD];
    if (half == 0) {
      const float z = 0.5f + (float)s * (1.5f / 64.0f);
      const float* o  = rays_o + (size_t)(b * RAYS + r) * 3;
      const float* dd = rays_d + (size_t)(b * RAYS + r) * 3;
      const float px = o[0] + dd[0] * z, py = o[1] + dd[1] * z, pz = o[2] + dd[2] * z;
      const float* T = transforms + (size_t)vox_idx[p] * 12;
      const float cx = T[0]*px + T[1]*py + T[2]*pz  + T[3];
      const float cy = T[4]*px + T[5]*py + T[6]*pz  + T[7];
      const float cz = T[8]*px + T[9]*py + T[10]*pz + T[11];
      row[0] = (_Float16)cx; row[1] = (_Float16)cy; row[2] = (_Float16)cz;
      #pragma unroll
      for (int l = 0; l < 10; ++l) {
        const float f = (float)(1 << l);
        const float xs = cx * f, ys = cy * f, zs = cz * f;
        row[3 + 6*l + 0] = (_Float16)__sinf(xs);
        row[3 + 6*l + 1] = (_Float16)__sinf(ys);
        row[3 + 6*l + 2] = (_Float16)__sinf(zs);
        row[3 + 6*l + 3] = (_Float16)__cosf(xs);
        row[3 + 6*l + 4] = (_Float16)__cosf(ys);
        row[3 + 6*l + 5] = (_Float16)__cosf(zs);
      }
      #pragma unroll
      for (int c = 211; c < K0PAD; ++c) row[c] = (_Float16)0.0f;
    } else {
      const float* src = uv_feat + ((size_t)b * UVS + (size_t)uv_idx[p]) * FEATC;
      #pragma unroll 4
      for (int c4 = 0; c4 < FEATC / 4; ++c4) {
        v4f v = *(const v4f*)(src + 4 * c4);
        row[63 + 4*c4 + 0] = (_Float16)v.x;
        row[63 + 4*c4 + 1] = (_Float16)v.y;
        row[63 + 4*c4 + 2] = (_Float16)v.z;
        row[63 + 4*c4 + 3] = (_Float16)v.w;
      }
    }
  }
  __syncthreads();

  const v8f vzero = {0.f,0.f,0.f,0.f,0.f,0.f,0.f,0.f};
  v8f acc0[8], acc1[8];

  // ---- layer 0: [64,224] @ [224,256] ----
  #pragma unroll
  for (int i = 0; i < 8; ++i) { acc0[i] = vzero; acc1[i] = vzero; }
  {
    const _Float16* ar0 = &smem[(mbase + lm) * K0PAD + lg * 16];
    const _Float16* ar1 = ar0 + 16 * K0PAD;
    const _Float16* wb  = W0T + (size_t)(nbase + lm) * K0PAD + lg * 16;
    mlp_layer<7, K0PAD>(ar0, ar1, wb, acc0, acc1);
  }
  __syncthreads();
  #pragma unroll
  for (int nt = 0; nt < 8; ++nt) {
    const int n = nbase + nt * 16 + lm;
    const float bb = bias0[n];
    #pragma unroll
    for (int v = 0; v < 8; ++v) {
      const int m = lg * 8 + v;
      smem[(mbase + m) * HID + n]      = (_Float16)fmaxf(acc0[nt][v] + bb, 0.0f);
      smem[(mbase + 16 + m) * HID + n] = (_Float16)fmaxf(acc1[nt][v] + bb, 0.0f);
    }
  }
  __syncthreads();

  // ---- layer 1: [64,256] @ [256,256] ----
  #pragma unroll
  for (int i = 0; i < 8; ++i) { acc0[i] = vzero; acc1[i] = vzero; }
  {
    const _Float16* ar0 = &smem[(mbase + lm) * HID + lg * 16];
    const _Float16* ar1 = ar0 + 16 * HID;
    const _Float16* wb  = W1T + (size_t)(nbase + lm) * HID + lg * 16;
    mlp_layer<8, HID>(ar0, ar1, wb, acc0, acc1);
  }
  __syncthreads();
  #pragma unroll
  for (int nt = 0; nt < 8; ++nt) {
    const int n = nbase + nt * 16 + lm;
    const float bb = bias1[n];
    #pragma unroll
    for (int v = 0; v < 8; ++v) {
      const int m = lg * 8 + v;
      smem[(mbase + m) * HID + n]      = (_Float16)fmaxf(acc0[nt][v] + bb, 0.0f);
      smem[(mbase + 16 + m) * HID + n] = (_Float16)fmaxf(acc1[nt][v] + bb, 0.0f);
    }
  }
  __syncthreads();

  // ---- layer 2: [64,256] @ [256,16(pad of 4)] ; each wave takes 16 rows ----
  v8f acc3 = vzero;
  {
    const _Float16* arow = &smem[(wave * 16 + lm) * HID + lg * 16];
    #pragma unroll
    for (int kk = 0; kk < 8; ++kk) {
      const v16h a  = load_afrag(arow, kk * 32);
      const v16h bf = load_bfrag(WoutT + (size_t)lm * HID + kk * 32 + lg * 16);
      acc3 = wmma_f16(a, bf, acc3);
    }
  }
  if (lm < 4) {
    const float bo = b_out[lm];
    #pragma unroll
    for (int v = 0; v < 8; ++v) {
      const int m = wave * 16 + lg * 8 + v;
      out4[(size_t)(base + m) * 4 + lm] = acc3[v] + bo;
    }
  }
}

// ---------------------------------------------------------------------------
// Kernel 2: per-ray alpha compositing (sequential N=64 scan).
// ---------------------------------------------------------------------------
__global__ __launch_bounds__(256)
void composite_kernel(const float* __restrict__ rays_d,
                      const float* __restrict__ out4,
                      float* __restrict__ rgb_map)
{
  const int idx = blockIdx.x * 256 + threadIdx.x;
  if (idx >= BATCH * RAYS) return;
  const float* dd = rays_d + (size_t)idx * 3;
  const float nrm = sqrtf(dd[0]*dd[0] + dd[1]*dd[1] + dd[2]*dd[2]);
  const float dist = 1.5f / 64.0f;
  float T = 1.0f, r = 0.f, g = 0.f, bl = 0.f;
  for (int s = 0; s < NSAMP; ++s) {
    v4f raw = *(const v4f*)(out4 + ((size_t)idx * NSAMP + s) * 4);
    const float sigma = fmaxf(raw.w, 0.0f);
    const float d = (s == NSAMP - 1) ? 1e10f : dist;
    const float alpha = 1.0f - __expf(-sigma * d * nrm);
    const float w = alpha * T;
    r  += w / (1.0f + __expf(-raw.x));
    g  += w / (1.0f + __expf(-raw.y));
    bl += w / (1.0f + __expf(-raw.z));
    T *= (1.0f - alpha + 1e-10f);
  }
  rgb_map[(size_t)idx * 3 + 0] = r;
  rgb_map[(size_t)idx * 3 + 1] = g;
  rgb_map[(size_t)idx * 3 + 2] = bl;
}

// ---------------------------------------------------------------------------
extern "C" void kernel_launch(void* const* d_in, const int* in_sizes, int n_in,
                              void* d_out, int out_size, void* d_ws, size_t ws_size,
                              hipStream_t stream) {
  const float* rays_o     = (const float*)d_in[0];
  const float* rays_d     = (const float*)d_in[1];
  const float* uv_feat    = (const float*)d_in[2];
  const float* transforms = (const float*)d_in[3];
  const float* w0         = (const float*)d_in[4];
  const float* b0         = (const float*)d_in[5];
  const float* w1         = (const float*)d_in[6];
  const float* b1         = (const float*)d_in[7];
  const float* w_out      = (const float*)d_in[8];
  const float* b_out      = (const float*)d_in[9];
  const int*   uv_idx     = (const int*)d_in[10];
  const int*   vox_idx    = (const int*)d_in[11];

  char* ws = (char*)d_ws;
  float*    out4  = (float*)ws;                                   // 16 MB raw [P][4]
  _Float16* W0T   = (_Float16*)(ws + (size_t)NPTS * 4 * sizeof(float));
  _Float16* W1T   = W0T + 256 * K0PAD;
  _Float16* WoutT = W1T + 256 * 256;

  const int wtot = 256 * K0PAD + 256 * 256 + 16 * 256;            // 126976
  prep_weights<<<(wtot + 255) / 256, 256, 0, stream>>>(w0, w1, w_out, W0T, W1T, WoutT);

  mlp_kernel<<<NPTS / MTILE, 128, 0, stream>>>(
      rays_o, rays_d, uv_feat, transforms, b0, b1, b_out,
      uv_idx, vox_idx, W0T, W1T, WoutT, out4);

  composite_kernel<<<(BATCH * RAYS) / 256, 256, 0, stream>>>(
      rays_d, out4, (float*)d_out);
}
